// VecKM_82970178224385
// MI455X (gfx1250) — matrix operand order repaired and checked by reading, not compiled, not run on hardware
//
#include <hip/hip_runtime.h>

typedef __attribute__((ext_vector_type(16))) _Float16 v16h;
typedef __attribute__((ext_vector_type(8)))  _Float16 v8h;
typedef __attribute__((ext_vector_type(8)))  float    v8f;

#define BSZ   2
#define NPTS  4096
#define DD    1024
#define WDIM  2048
#define NA    2
#define NB    2

// ---------------- workspace layout (units: _Float16 elements) ----------------
constexpr size_t SZ_WMAT = (size_t)BSZ * NB * (size_t)WDIM * NPTS;   // 33,554,432
constexpr size_t SZ_TMAT = (size_t)BSZ * NA * (size_t)DD   * NPTS;   // 16,777,216
constexpr size_t SZ_K    = (size_t)BSZ * NB * NA * (size_t)DD * WDIM;// 16,777,216
constexpr size_t OFF_CWT = 0;                      // [bs][b][w][n]
constexpr size_t OFF_SWT = OFF_CWT + SZ_WMAT;
constexpr size_t OFF_CTT = OFF_SWT + SZ_WMAT;      // [bs][a][d][n]
constexpr size_t OFF_STT = OFF_CTT + SZ_TMAT;
constexpr size_t OFF_CW  = OFF_STT + SZ_TMAT;      // [bs][b][n][w]
constexpr size_t OFF_SW  = OFF_CW  + SZ_WMAT;
constexpr size_t OFF_KRE = OFF_SW  + SZ_WMAT;      // [bs][b][a][d][w]  (f16)
constexpr size_t OFF_KIM = OFF_KRE + SZ_K;
// total = 201,326,592 f16 = ~403 MB of d_ws

// ---------------- WMMA helpers ----------------
__device__ __forceinline__ v8f wmma16(v16h a, v16h b, v8f c) {
  // D = A(16x32 f16) * B(32x16 f16) + C(16x16 f32)
  return __builtin_amdgcn_wmma_f32_16x16x32_f16(false, a, false, b, (short)0, c,
                                                false, false);
}

// A fragment: lane holds row m=lq; 16 halves = k: {h*8..h*8+7} and {16+h*8..16+h*8+7}
__device__ __forceinline__ v16h loadA(const _Float16* rowptr, int kk, int half) {
  union { v16h v; v8h h[2]; } u;
  u.h[0] = *(const v8h*)(rowptr + kk + half * 8);
  u.h[1] = *(const v8h*)(rowptr + kk + 16 + half * 8);
  return u.v;
}

// B fragment: lane holds column n=lq; 16 halves = k: {h*8..h*8+7} and {16+h*8..16+h*8+7}
__device__ __forceinline__ v16h loadB(const _Float16* colptr, int kk, int half) {
  union { v16h v; v8h h[2]; } u;
  u.h[0] = *(const v8h*)(colptr + kk + half * 8);
  u.h[1] = *(const v8h*)(colptr + kk + 16 + half * 8);
  return u.v;
}

// ---------------- phase (cos/sin) table kernels ----------------
// Cw/Sw [bs][b][n][w]  (w fastest)
__global__ __launch_bounds__(256) void phaseW_row(const float* __restrict__ pts,
                                                  const float* __restrict__ W,
                                                  _Float16* __restrict__ cw,
                                                  _Float16* __restrict__ sw) {
  size_t idx = (size_t)blockIdx.x * 256 + threadIdx.x;
  int w = (int)(idx % WDIM); size_t t = idx / WDIM;
  int n = (int)(t % NPTS);   t /= NPTS;
  int b = (int)(t % NB);     int bs = (int)(t / NB);
  const float* p  = pts + ((size_t)bs * NPTS + n) * 3;
  const float* Wb = W + (size_t)b * 3 * WDIM;
  float ph = p[0] * Wb[w] + p[1] * Wb[WDIM + w] + p[2] * Wb[2 * WDIM + w];
  float s, c; __sincosf(ph, &s, &c);
  cw[idx] = (_Float16)c; sw[idx] = (_Float16)s;
}

// CwT/SwT [bs][b][w][n]  (n fastest)
__global__ __launch_bounds__(256) void phaseW_col(const float* __restrict__ pts,
                                                  const float* __restrict__ W,
                                                  _Float16* __restrict__ cwt,
                                                  _Float16* __restrict__ swt) {
  size_t idx = (size_t)blockIdx.x * 256 + threadIdx.x;
  int n = (int)(idx % NPTS); size_t t = idx / NPTS;
  int w = (int)(t % WDIM);   t /= WDIM;
  int b = (int)(t % NB);     int bs = (int)(t / NB);
  const float* p  = pts + ((size_t)bs * NPTS + n) * 3;
  const float* Wb = W + (size_t)b * 3 * WDIM;
  float ph = p[0] * Wb[w] + p[1] * Wb[WDIM + w] + p[2] * Wb[2 * WDIM + w];
  float s, c; __sincosf(ph, &s, &c);
  cwt[idx] = (_Float16)c; swt[idx] = (_Float16)s;
}

// CtT/StT [bs][a][d][n]  (n fastest)
__global__ __launch_bounds__(256) void phaseT_col(const float* __restrict__ pts,
                                                  const float* __restrict__ T,
                                                  _Float16* __restrict__ ctt,
                                                  _Float16* __restrict__ stt) {
  size_t idx = (size_t)blockIdx.x * 256 + threadIdx.x;
  int n = (int)(idx % NPTS); size_t t = idx / NPTS;
  int d = (int)(t % DD);     t /= DD;
  int a = (int)(t % NA);     int bs = (int)(t / NA);
  const float* p  = pts + ((size_t)bs * NPTS + n) * 3;
  const float* Ta = T + (size_t)a * 3 * DD;
  float ph = p[0] * Ta[d] + p[1] * Ta[DD + d] + p[2] * Ta[2 * DD + d];
  float s, c; __sincosf(ph, &s, &c);
  ctt[idx] = (_Float16)c; stt[idx] = (_Float16)s;
}

// ---------------- stage A: K[w,d] = sum_n conj(eW[n,w]) * eT[n,d] ----------------
// M axis = d (rows), column axis = w  => C-tile stores to K[d][w] are lane-coalesced.
// Kre = sum ct*cw + st*sw ; Kim = sum st*cw + (-ct)*sw  (negate A-side: 2 frags only)
__global__ __launch_bounds__(32) void gemmK_kernel(const _Float16* __restrict__ ws,
                                                   _Float16* __restrict__ kre,
                                                   _Float16* __restrict__ kim) {
  const int lane = threadIdx.x & 31;
  const int half = lane >> 4, lq = lane & 15;
  const int w0 = blockIdx.x * 64;
  const int d0 = blockIdx.y * 32;
  const int z  = blockIdx.z;            // ((bs*NB)+b)*NA + a
  const int a  = z & 1;
  const int b  = (z >> 1) & 1;
  const int bs = z >> 2;

  const _Float16* CtT = ws + OFF_CTT + (size_t)(bs * NA + a) * DD * NPTS;
  const _Float16* StT = ws + OFF_STT + (size_t)(bs * NA + a) * DD * NPTS;
  const _Float16* CwT = ws + OFF_CWT + (size_t)(bs * NB + b) * WDIM * NPTS;
  const _Float16* SwT = ws + OFF_SWT + (size_t)(bs * NB + b) * WDIM * NPTS;

  v8f accRe[2][4] = {};   // + ct*cw + st*sw
  v8f accIm[2][4] = {};   // + st*cw - ct*sw

  for (int kk = 0; kk < NPTS; kk += 32) {
    v16h act[2], ast[2], nact[2];
#pragma unroll
    for (int dt = 0; dt < 2; ++dt) {
      const _Float16* rc = CtT + (size_t)(d0 + dt * 16 + lq) * NPTS;
      const _Float16* rs = StT + (size_t)(d0 + dt * 16 + lq) * NPTS;
      __builtin_prefetch(rc + kk + 64, 0, 3);
      act[dt]  = loadA(rc, kk, half);
      ast[dt]  = loadA(rs, kk, half);
      nact[dt] = -act[dt];
    }
    v16h bcw[4], bsw[4];
#pragma unroll
    for (int wt = 0; wt < 4; ++wt) {
      const _Float16* cc = CwT + (size_t)(w0 + wt * 16 + lq) * NPTS;
      const _Float16* cs = SwT + (size_t)(w0 + wt * 16 + lq) * NPTS;
      __builtin_prefetch(cc + kk + 64, 0, 3);
      bcw[wt] = loadB(cc, kk, half);
      bsw[wt] = loadB(cs, kk, half);
    }
#pragma unroll
    for (int dt = 0; dt < 2; ++dt) {
#pragma unroll
      for (int wt = 0; wt < 4; ++wt) {
        accRe[dt][wt] = wmma16(act[dt],  bcw[wt], accRe[dt][wt]);  //  ct*cw
        accRe[dt][wt] = wmma16(ast[dt],  bsw[wt], accRe[dt][wt]);  //  st*sw
        accIm[dt][wt] = wmma16(ast[dt],  bcw[wt], accIm[dt][wt]);  //  st*cw
        accIm[dt][wt] = wmma16(nact[dt], bsw[wt], accIm[dt][wt]);  // -ct*sw
      }
    }
  }

  _Float16* outRe = kre + (size_t)z * DD * WDIM;
  _Float16* outIm = kim + (size_t)z * DD * WDIM;
#pragma unroll
  for (int dt = 0; dt < 2; ++dt) {
#pragma unroll
    for (int wt = 0; wt < 4; ++wt) {
#pragma unroll
      for (int r = 0; r < 8; ++r) {
        int d = d0 + dt * 16 + r + half * 8;
        int w = w0 + wt * 16 + lq;
        outRe[(size_t)d * WDIM + w] = (_Float16)accRe[dt][wt][r];
        outIm[(size_t)d * WDIM + w] = (_Float16)accIm[dt][wt][r];
      }
    }
  }
}

// ---------------- stage B: G = eW*K, /eT, row-normalize, *32, write complex64 ------
// Block: 256 threads (8 waves), 16 points x full D=1024. Wave owns 16x128 (8 tiles).
// Gre = sum cw*Kre + (-sw)*Kim ; Gim = sum cw*Kim + sw*Kre  (negate A-side: 1 frag)
__global__ __launch_bounds__(256) void gemmG_kernel(const _Float16* __restrict__ ws,
                                                    const _Float16* __restrict__ kre,
                                                    const _Float16* __restrict__ kim,
                                                    const float* __restrict__ pts,
                                                    const float* __restrict__ T,
                                                    float* __restrict__ out) {
  __shared__ float pbuf[48];
  __shared__ float nrm[16];
  const int tid  = threadIdx.x;
  const int lane = tid & 31, wid = tid >> 5;
  const int half = lane >> 4, lq = lane & 15;
  const int n0 = blockIdx.x * 16;
  const int z  = blockIdx.y;
  const int a  = z & 1;
  const int b  = (z >> 1) & 1;
  const int bs = z >> 2;

  if (tid < 16) nrm[tid] = 0.0f;
  if (tid < 48) pbuf[tid] = pts[((size_t)bs * NPTS + n0) * 3 + tid];
  __syncthreads();

  const _Float16* Cw  = ws + OFF_CW + (size_t)(bs * NB + b) * NPTS * WDIM;
  const _Float16* Sw  = ws + OFF_SW + (size_t)(bs * NB + b) * NPTS * WDIM;
  const _Float16* Kre = kre + (size_t)z * DD * WDIM;
  const _Float16* Kim = kim + (size_t)z * DD * WDIM;
  const int c0 = wid * 128;

  const _Float16* arow_c = Cw + (size_t)(n0 + lq) * WDIM;
  const _Float16* arow_s = Sw + (size_t)(n0 + lq) * WDIM;

  v8f gre[8] = {};
  v8f gim[8] = {};

  for (int kk = 0; kk < WDIM; kk += 32) {
    __builtin_prefetch(arow_c + kk + 64, 0, 3);
    v16h acw  = loadA(arow_c, kk, half);
    v16h asw  = loadA(arow_s, kk, half);
    v16h nasw = -asw;
#pragma unroll
    for (int dt = 0; dt < 8; ++dt) {
      const _Float16* colr = Kre + (size_t)(c0 + dt * 16 + lq) * WDIM;
      const _Float16* coli = Kim + (size_t)(c0 + dt * 16 + lq) * WDIM;
      v16h bre = loadB(colr, kk, half);
      v16h bim = loadB(coli, kk, half);
      gre[dt] = wmma16(acw,  bre, gre[dt]);   //  cw*Kre
      gre[dt] = wmma16(nasw, bim, gre[dt]);   // -sw*Kim
      gim[dt] = wmma16(acw,  bim, gim[dt]);   //  cw*Kim
      gim[dt] = wmma16(asw,  bre, gim[dt]);   //  sw*Kre
    }
  }

  // Row norms: |G/eT| == |G| since |eT|==1. Reduce 16 lanes per half, then across waves.
#pragma unroll
  for (int r = 0; r < 8; ++r) {
    float v = 0.0f;
#pragma unroll
    for (int dt = 0; dt < 8; ++dt)
      v += gre[dt][r] * gre[dt][r] + gim[dt][r] * gim[dt][r];
#pragma unroll
    for (int m = 1; m < 16; m <<= 1) v += __shfl_xor(v, m, 32);
    if (lq == 0) atomicAdd(&nrm[r + half * 8], v);
  }
  __syncthreads();

  // Epilogue: recompute eT, multiply by conj (== divide, |eT|=1), scale, store.
  const float* Ta = T + (size_t)a * 3 * DD;
  float2* o = (float2*)out + ((size_t)z * NPTS + n0) * DD;
#pragma unroll
  for (int r = 0; r < 8; ++r) {
    const int row = r + half * 8;
    const float scale = 32.0f * rsqrtf(nrm[row]);
    const float px = pbuf[row * 3 + 0];
    const float py = pbuf[row * 3 + 1];
    const float pz = pbuf[row * 3 + 2];
#pragma unroll
    for (int dt = 0; dt < 8; ++dt) {
      const int d = c0 + dt * 16 + lq;
      float ph = px * Ta[d] + py * Ta[DD + d] + pz * Ta[2 * DD + d];
      float s, c; __sincosf(ph, &s, &c);
      const float gr = gre[dt][r];
      const float gi = gim[dt][r];
      float2 res;
      res.x = (gr * c + gi * s) * scale;
      res.y = (gi * c - gr * s) * scale;
      o[(size_t)row * DD + d] = res;
    }
  }
}

// ---------------- host launcher ----------------
extern "C" void kernel_launch(void* const* d_in, const int* in_sizes, int n_in,
                              void* d_out, int out_size, void* d_ws, size_t ws_size,
                              hipStream_t stream) {
  const float* pts = (const float*)d_in[0];   // (BS, N, 3)
  const float* T   = (const float*)d_in[1];   // (NA, 3, D)
  const float* W   = (const float*)d_in[2];   // (NB, 3, W_DIM)
  float* out = (float*)d_out;                 // complex64 interleaved (BS,4,N,D)
  _Float16* ws = (_Float16*)d_ws;

  {
    const size_t n = (size_t)BSZ * NB * NPTS * WDIM;
    phaseW_row<<<dim3((unsigned)(n / 256)), 256, 0, stream>>>(pts, W, ws + OFF_CW, ws + OFF_SW);
    phaseW_col<<<dim3((unsigned)(n / 256)), 256, 0, stream>>>(pts, W, ws + OFF_CWT, ws + OFF_SWT);
  }
  {
    const size_t n = (size_t)BSZ * NA * DD * NPTS;
    phaseT_col<<<dim3((unsigned)(n / 256)), 256, 0, stream>>>(pts, T, ws + OFF_CTT, ws + OFF_STT);
  }
  gemmK_kernel<<<dim3(WDIM / 64, DD / 32, BSZ * NB * NA), 32, 0, stream>>>(
      ws, ws + OFF_KRE, ws + OFF_KIM);
  gemmG_kernel<<<dim3(NPTS / 16, BSZ * NB * NA), 256, 0, stream>>>(
      ws, ws + OFF_KRE, ws + OFF_KIM, pts, T, out);
}
